// RelativeAttention_83519934038568
// MI455X (gfx1250) — matrix operand order repaired and checked by reading, not compiled, not run on hardware
//
#include <hip/hip_runtime.h>
#include <hip/hip_fp16.h>
#include <math.h>

typedef __attribute__((ext_vector_type(16))) _Float16 v16h;
typedef __attribute__((ext_vector_type(8)))  float    v8f;
typedef __attribute__((ext_vector_type(4)))  _Float16 v4h;

#define K_DIM   1024   // feature dim
#define N_ANCH  4096   // anchors
#define H_DIM   2048   // output hidden
#define M_TILE  32     // rows per workgroup (2 sub-tiles of 16)
#define THREADS 512    // 16 waves
#define NWAVES  16
#define BLK     256    // anchors per block iteration (16 waves x 16)
#define NBLK    (N_ANCH / BLK)   // 16
#define HPW     128    // output columns per wave (16 x 128 = 2048)
#define NT      (HPW / 16)       // 8 N-tiles per wave

// LDS partition (dynamic): A tile 64KB | P tile 16KB | rowsum 128B
#define ALDS_BYTES (M_TILE * K_DIM * 2)   // 65536
#define PLDS_BYTES (M_TILE * BLK * 2)     // 16384
#define SMEM_BYTES (ALDS_BYTES + PLDS_BYTES + M_TILE * 4)

// ---------------------------------------------------------------------------
// L2-normalize each 1024-float row, emit f16. One 256-thread block per row.
// ---------------------------------------------------------------------------
__global__ void l2norm_rows_f16(const float* __restrict__ in,
                                _Float16* __restrict__ out) {
  const int row = blockIdx.x;
  const int tid = threadIdx.x;
  const float4* r4 = reinterpret_cast<const float4*>(in + (size_t)row * K_DIM);
  float4 v = r4[tid];
  float ss = v.x * v.x + v.y * v.y + v.z * v.z + v.w * v.w;
#pragma unroll
  for (int off = 16; off > 0; off >>= 1) ss += __shfl_xor(ss, off, 32);
  __shared__ float ws[8];
  const int wave = tid >> 5, lane = tid & 31;
  if (lane == 0) ws[wave] = ss;
  __syncthreads();
  float tot = 0.0f;
#pragma unroll
  for (int i = 0; i < 8; i++) tot += ws[i];
  const float scale = 1.0f / fmaxf(sqrtf(tot), 1e-12f);
  v4h o;
  o[0] = (_Float16)(v.x * scale);
  o[1] = (_Float16)(v.y * scale);
  o[2] = (_Float16)(v.z * scale);
  o[3] = (_Float16)(v.w * scale);
  *reinterpret_cast<v4h*>(out + (size_t)row * K_DIM + tid * 4) = o;
}

// ---------------------------------------------------------------------------
// values [4096][2048] f32 -> vT [2048][4096] f16  (K-contiguous B fragments)
// ---------------------------------------------------------------------------
__global__ void transpose_f32_f16(const float* __restrict__ v,
                                  _Float16* __restrict__ vt) {
  __shared__ float tile[32][33];
  const int c0 = blockIdx.x * 32;  // H dim
  const int r0 = blockIdx.y * 32;  // anchor dim
  for (int i = threadIdx.y; i < 32; i += 8)
    tile[i][threadIdx.x] = v[(size_t)(r0 + i) * H_DIM + c0 + threadIdx.x];
  __syncthreads();
  for (int i = threadIdx.y; i < 32; i += 8)
    vt[(size_t)(c0 + i) * N_ANCH + r0 + threadIdx.x] =
        (_Float16)tile[threadIdx.x][i];
}

// ---------------------------------------------------------------------------
// Fused: sims (WMMA f16) -> quantize -> exp -> P@V (WMMA f16) -> /rowsum.
// 512 threads (16 waves) per workgroup: 32 rows x full 2048 output.
// Each wave: 2 M-subtiles sharing every B fragment (halves L2 B traffic).
// ---------------------------------------------------------------------------
__global__ __launch_bounds__(THREADS, 1)
void fused_relattn(const _Float16* __restrict__ xh,
                   const _Float16* __restrict__ ah,
                   const _Float16* __restrict__ vT,
                   float* __restrict__ out) {
  extern __shared__ char smem[];
  _Float16* Alds = (_Float16*)smem;                       // 32 x 1024 f16
  _Float16* Plds = (_Float16*)(smem + ALDS_BYTES);        // 32 x 256  f16
  float* rowsum  = (float*)(smem + ALDS_BYTES + PLDS_BYTES);

  const int tid   = threadIdx.x;
  const int wave  = tid >> 5;
  const int lane  = tid & 31;
  const int lmod  = lane & 15;
  const int lhalf = lane >> 4;  // 0: lanes 0-15, 1: lanes 16-31
  const size_t mbase = (size_t)blockIdx.x * M_TILE;

  // ---- Stage the 32x1024 A tile into LDS with async global->LDS copies ----
  {
    const uint64_t gsrc = (uint64_t)(uintptr_t)(xh + mbase * K_DIM);
    const uint32_t lbase = (uint32_t)(uintptr_t)Alds;  // LDS offset = low 32b
#pragma unroll
    for (int i = 0; i < ALDS_BYTES / 16 / THREADS; i++) {  // 8 x 16B chunks
      const uint32_t off = (uint32_t)(tid + i * THREADS) * 16u;
      uint32_t loff = lbase + off;
      uint64_t ga = gsrc + off;
      asm volatile("global_load_async_to_lds_b128 %0, %1, off"
                   :: "v"(loff), "v"(ga) : "memory");
    }
    asm volatile("s_wait_asynccnt 0x0" ::: "memory");
  }
  if (tid < M_TILE) rowsum[tid] = 0.0f;
  __syncthreads();

  v8f acc[2][NT];  // [msub][ntile] f32 C-layout accumulators (128 VGPRs)
#pragma unroll
  for (int m = 0; m < 2; m++)
#pragma unroll
    for (int t = 0; t < NT; t++)
#pragma unroll
      for (int i = 0; i < 8; i++) acc[m][t][i] = 0.0f;
  float sum8[2][8];  // per-row softmax denominators (C layout)
#pragma unroll
  for (int m = 0; m < 2; m++)
#pragma unroll
    for (int i = 0; i < 8; i++) sum8[m][i] = 0.0f;

  const _Float16* arow0 = &Alds[lmod * K_DIM];         // rows 0..15
  const _Float16* arow1 = &Alds[(16 + lmod) * K_DIM];  // rows 16..31

  for (int blk = 0; blk < NBLK; blk++) {
    // ---- GEMM1: two 16x16 similarity tiles, shared B fragment ----
    v8f s0, s1;
#pragma unroll
    for (int i = 0; i < 8; i++) { s0[i] = 0.0f; s1[i] = 0.0f; }
    const int aidx = blk * BLK + wave * 16 + lmod;  // B frag: lane col = lmod
    const _Float16* brow = ah + (size_t)aidx * K_DIM;
#pragma unroll 4
    for (int ks = 0; ks < K_DIM / 32; ks++) {
      const int kb = ks * 32;
      v16h a0, a1, b;
      // B 32x16 f16: half0 K 0-7 & 16-23, half1 K 8-15 & 24-31
      reinterpret_cast<uint4*>(&b)[0] =
          *reinterpret_cast<const uint4*>(brow + kb + lhalf * 8);
      reinterpret_cast<uint4*>(&b)[1] =
          *reinterpret_cast<const uint4*>(brow + kb + 16 + lhalf * 8);
      // A 16x32 f16: half0 K 0-7 & 16-23, half1 K 8-15 & 24-31
      reinterpret_cast<uint4*>(&a0)[0] =
          *reinterpret_cast<const uint4*>(arow0 + kb + lhalf * 8);
      reinterpret_cast<uint4*>(&a0)[1] =
          *reinterpret_cast<const uint4*>(arow0 + kb + 16 + lhalf * 8);
      reinterpret_cast<uint4*>(&a1)[0] =
          *reinterpret_cast<const uint4*>(arow1 + kb + lhalf * 8);
      reinterpret_cast<uint4*>(&a1)[1] =
          *reinterpret_cast<const uint4*>(arow1 + kb + 16 + lhalf * 8);
      s0 = __builtin_amdgcn_wmma_f32_16x16x32_f16(false, a0, false, b,
                                                  (short)0, s0, false, false);
      s1 = __builtin_amdgcn_wmma_f32_16x16x32_f16(false, a1, false, b,
                                                  (short)0, s1, false, false);
    }

    __syncthreads();  // previous GEMM2 readers are done with Plds
    // quantize (v_rndne = round-half-even, matches jnp.round) -> exp.
    // |sim| <= 1 so softmax needs no max subtraction.
#pragma unroll
    for (int r = 0; r < 8; r++) {
      const float q0 = rintf(s0[r] * 20.0f) * 0.05f;
      const float q1 = rintf(s1[r] * 20.0f) * 0.05f;
      const float e0 = __expf(q0);
      const float e1 = __expf(q1);
      sum8[0][r] += e0;
      sum8[1][r] += e1;
      Plds[(r + 8 * lhalf) * BLK + wave * 16 + lmod] = (_Float16)e0;
      Plds[(16 + r + 8 * lhalf) * BLK + wave * 16 + lmod] = (_Float16)e1;
    }
    __syncthreads();

    // ---- GEMM2 partial: P(32xBLK) x V(BLK x 128-col slice) ----
#pragma unroll
    for (int k2 = 0; k2 < BLK / 32; k2++) {  // 8 K-steps of 32
      v16h a20, a21;
      {
        const _Float16* prow0 = &Plds[lmod * BLK + k2 * 32];
        const _Float16* prow1 = &Plds[(16 + lmod) * BLK + k2 * 32];
        reinterpret_cast<uint4*>(&a20)[0] =
            *reinterpret_cast<const uint4*>(prow0 + lhalf * 8);
        reinterpret_cast<uint4*>(&a20)[1] =
            *reinterpret_cast<const uint4*>(prow0 + 16 + lhalf * 8);
        reinterpret_cast<uint4*>(&a21)[0] =
            *reinterpret_cast<const uint4*>(prow1 + lhalf * 8);
        reinterpret_cast<uint4*>(&a21)[1] =
            *reinterpret_cast<const uint4*>(prow1 + 16 + lhalf * 8);
      }
#pragma unroll
      for (int t = 0; t < NT; t++) {
        const int n = wave * HPW + t * 16 + lmod;
        const _Float16* vrow = vT + (size_t)n * N_ANCH + blk * BLK;
        if (k2 == 0 && blk + 1 < NBLK)
          __builtin_prefetch(vrow + BLK, 0, 3);  // next block's B chunk
        v16h b2;
        reinterpret_cast<uint4*>(&b2)[0] =
            *reinterpret_cast<const uint4*>(vrow + k2 * 32 + lhalf * 8);
        reinterpret_cast<uint4*>(&b2)[1] =
            *reinterpret_cast<const uint4*>(vrow + k2 * 32 + 16 + lhalf * 8);
        acc[0][t] = __builtin_amdgcn_wmma_f32_16x16x32_f16(
            false, a20, false, b2, (short)0, acc[0][t], false, false);
        acc[1][t] = __builtin_amdgcn_wmma_f32_16x16x32_f16(
            false, a21, false, b2, (short)0, acc[1][t], false, false);
      }
    }
  }

  // ---- softmax denominators: reduce over N (16 lanes of each half) ----
#pragma unroll
  for (int m = 0; m < 2; m++)
#pragma unroll
    for (int r = 0; r < 8; r++) {
#pragma unroll
      for (int off = 1; off < 16; off <<= 1)
        sum8[m][r] += __shfl_xor(sum8[m][r], off, 32);
    }
  if (lmod == 0) {
#pragma unroll
    for (int m = 0; m < 2; m++)
#pragma unroll
      for (int r = 0; r < 8; r++)
        atomicAdd(&rowsum[m * 16 + r + 8 * lhalf], sum8[m][r]);  // ds_add_f32
  }
  __syncthreads();
  float rinv[2][8];
#pragma unroll
  for (int m = 0; m < 2; m++)
#pragma unroll
    for (int r = 0; r < 8; r++)
      rinv[m][r] = 1.0f / rowsum[m * 16 + r + 8 * lhalf];

  // ---- scale + store (lanes 0-15 per tile are N-contiguous: coalesced) ----
#pragma unroll
  for (int m = 0; m < 2; m++)
#pragma unroll
    for (int t = 0; t < NT; t++)
#pragma unroll
      for (int r = 0; r < 8; r++) {
        const size_t row = mbase + m * 16 + r + 8 * lhalf;
        out[row * H_DIM + wave * HPW + t * 16 + lmod] =
            acc[m][t][r] * rinv[m][r];
      }
}

// ---------------------------------------------------------------------------
extern "C" void kernel_launch(void* const* d_in, const int* in_sizes, int n_in,
                              void* d_out, int out_size, void* d_ws,
                              size_t ws_size, hipStream_t stream) {
  const float* x       = (const float*)d_in[0];  // [16384,1024]
  const float* anchors = (const float*)d_in[1];  // [4096,1024]
  const float* values  = (const float*)d_in[2];  // [4096,2048]
  float* out = (float*)d_out;                    // [16384,2048]

  char* ws = (char*)d_ws;
  _Float16* xh = (_Float16*)ws;                                   // 32 MB
  _Float16* ah = (_Float16*)(ws + (size_t)16384 * 1024 * 2);      //  8 MB
  _Float16* vT = (_Float16*)(ws + (size_t)16384 * 1024 * 2 +
                             (size_t)4096 * 1024 * 2);            // 16 MB

  l2norm_rows_f16<<<16384, 256, 0, stream>>>(x, xh);
  l2norm_rows_f16<<<4096, 256, 0, stream>>>(anchors, ah);
  transpose_f32_f16<<<dim3(H_DIM / 32, N_ANCH / 32), dim3(32, 8), 0, stream>>>(
      values, vT);
  fused_relattn<<<16384 / M_TILE, THREADS, SMEM_BYTES, stream>>>(xh, ah, vT,
                                                                 out);
}
